// GroupedFrequencyAttention_25400436589033
// MI455X (gfx1250) — matrix operand order repaired and checked by reading, not compiled, not run on hardware
//
#include <hip/hip_runtime.h>
#include <hip/hip_bf16.h>
#include <math.h>

// ---------------- problem constants ----------------
#define BB 32
#define GG 3
#define LL 1024
#define DD 384
#define HH 96
#define TT (BB*GG*LL)          // 98304 tokens
#define BL (BB*LL)
#define EPS 1e-5f

typedef __attribute__((ext_vector_type(16))) __bf16 v16bf;
typedef __attribute__((ext_vector_type(8)))  float  v8f;

__device__ __forceinline__ v8f wmma_bf16(v16bf a, v16bf b, v8f c) {
  // D = A(16x32 bf16) * B(32x16 bf16) + C(16x16 f32)
  return __builtin_amdgcn_wmma_f32_16x16x32_bf16(false, a, false, b, (short)0, c, false, false);
}

// A fragment: 16x32 bf16 tile from row-major [rows][lda] bf16 matrix.
// ISA layout: lane m (0..15) rows; VGPR v<4 -> K = 2v, 2v+1 (+8*half); v>=4 -> K = 16+2(v-4) (+8*half)
__device__ __forceinline__ v16bf load_a(const __bf16* A, int lda, int row0, int k0) {
  int lane = threadIdx.x & 31;
  int m = lane & 15, half = lane >> 4;
  union { v16bf v; unsigned u[8]; } r;
  const __bf16* base = A + (size_t)(row0 + m) * lda + k0 + 8 * half;
#pragma unroll
  for (int v = 0; v < 4; ++v) {
    r.u[v]     = *(const unsigned*)(base + 2 * v);
    r.u[v + 4] = *(const unsigned*)(base + 16 + 2 * v);
  }
  return r.v;
}

// B fragment: 32x16 bf16 tile. Bt is transposed weights, row-major [N][K] (ldb = K stride).
// ISA layout: VGPR v<4 -> K = 2v, 2v+1 (+8*half); v>=4 -> K = 16+2(v-4) (+8*half); column n = lane&15
__device__ __forceinline__ v16bf load_b(const __bf16* Bt, int ldb, int k0, int n0) {
  int lane = threadIdx.x & 31;
  int n = lane & 15, half = lane >> 4;
  union { v16bf v; unsigned u[8]; } r;
  const __bf16* base = Bt + (size_t)(n0 + n) * ldb + k0 + 8 * half;
#pragma unroll
  for (int v = 0; v < 4; ++v) {
    r.u[v]     = *(const unsigned*)(base + 2 * v);
    r.u[v + 4] = *(const unsigned*)(base + 16 + 2 * v);
  }
  return r.v;
}

// A fragment built from fp32 LDS tile h[16][ldh]
__device__ __forceinline__ v16bf load_a_lds(const float* h, int ldh, int k0) {
  int lane = threadIdx.x & 31;
  int m = lane & 15, half = lane >> 4;
  union { v16bf v; unsigned short s[16]; } r;
#pragma unroll
  for (int v = 0; v < 8; ++v) {
    int koff = 8 * half + ((v < 4) ? 2 * v : 16 + 2 * (v - 4));
    union { __bf16 h2; unsigned short u; } a0, a1;
    a0.h2 = (__bf16)h[m * ldh + k0 + koff];
    a1.h2 = (__bf16)h[m * ldh + k0 + koff + 1];
    r.s[2 * v]     = a0.u;
    r.s[2 * v + 1] = a1.u;
  }
  return r.v;
}

// A fragment from bf16 LDS tile (row stride lda elements)
__device__ __forceinline__ v16bf load_a_ldsb(const __bf16* sA, int lda, int k0) {
  int lane = threadIdx.x & 31;
  int m = lane & 15, half = lane >> 4;
  union { v16bf v; unsigned u[8]; } r;
  const __bf16* base = sA + m * lda + k0 + 8 * half;
#pragma unroll
  for (int v = 0; v < 4; ++v) {
    r.u[v]     = *(const unsigned*)(base + 2 * v);
    r.u[v + 4] = *(const unsigned*)(base + 16 + 2 * v);
  }
  return r.v;
}

// ---------------- weight prep ----------------
__global__ __launch_bounds__(256) void k_transpose_bf16(const float* __restrict__ src,
                                                        __bf16* __restrict__ dst, int R, int C) {
  int i = blockIdx.x * 256 + threadIdx.x;
  if (i >= R * C) return;
  int r = i / C, c = i % C;
  dst[(size_t)c * R + r] = (__bf16)src[(size_t)r * C + c];
}

// conv_W (O=384, I=384, 3, 1) -> convWt[k][o][i] bf16
__global__ __launch_bounds__(256) void k_conv_w(const float* __restrict__ w, __bf16* __restrict__ dst) {
  int i = blockIdx.x * 256 + threadIdx.x;
  if (i >= 3 * DD * DD) return;
  int k = i / (DD * DD);
  int o = (i / DD) % DD;
  int in = i % DD;
  dst[i] = (__bf16)w[((size_t)o * DD + in) * 3 + k];
}

// ---------------- glob mean + bf16 casts ----------------
__global__ __launch_bounds__(256) void k_glob(const float* __restrict__ x,
                                              float* __restrict__ globF,
                                              __bf16* __restrict__ globB,
                                              __bf16* __restrict__ xB) {
  int i = blockIdx.x * 256 + threadIdx.x;
  if (i >= BL * DD) return;
  int d = i % DD;
  int l = (i / DD) % LL;
  int b = i / (DD * LL);
  const float* base = x + ((size_t)b * (GG * LL + 1) + 1) * DD;
  float v0 = base[((size_t)(0 * LL + l)) * DD + d];
  float v1 = base[((size_t)(1 * LL + l)) * DD + d];
  float v2 = base[((size_t)(2 * LL + l)) * DD + d];
  float m = (v0 + v1 + v2) * (1.0f / 3.0f);
  size_t gi = (size_t)(b * LL + l) * DD + d;
  globF[gi] = m;
  globB[gi] = (__bf16)m;
  xB[((size_t)((b * GG + 0) * LL + l)) * DD + d] = (__bf16)v0;
  xB[((size_t)((b * GG + 1) * LL + l)) * DD + d] = (__bf16)v1;
  xB[((size_t)((b * GG + 2) * LL + l)) * DD + d] = (__bf16)v2;
}

// ---------------- gate MLP (WMMA) + gating ----------------
__global__ __launch_bounds__(32) void k_gate(const float* __restrict__ x,
                                             const float* __restrict__ globF,
                                             const __bf16* __restrict__ xB,
                                             const __bf16* __restrict__ globB,
                                             const __bf16* __restrict__ cgW1t,
                                             const float* __restrict__ cg_b1,
                                             const float* __restrict__ cg_g,
                                             const float* __restrict__ cg_beta,
                                             const float* __restrict__ cg_W2,
                                             const float* __restrict__ cg_b2,
                                             __bf16* __restrict__ xgB) {
  __shared__ float h[16][192];
  __shared__ float gate[16];
  int t0 = blockIdx.x * 16;
  int b = t0 / (GG * LL);
  int l = t0 % LL;
  int grow0 = b * LL + l;
  int lane = threadIdx.x;

  v8f acc[12];
#pragma unroll
  for (int n = 0; n < 12; ++n) acc[n] = (v8f)0.0f;

  for (int k = 0; k < 24; ++k) {
    int k0 = k * 32;
    v16bf a = (k0 < DD) ? load_a(xB, DD, t0, k0)
                        : load_a(globB, DD, grow0, k0 - DD);
#pragma unroll
    for (int n = 0; n < 12; ++n) {
      v16bf bm = load_b(cgW1t, 2 * DD, k0, n * 16);
      acc[n] = wmma_bf16(a, bm, acc[n]);
    }
  }
  // store h = pre-LN hidden (+b1) to LDS
  {
    int nn = lane & 15, half = lane >> 4;
#pragma unroll
    for (int n = 0; n < 12; ++n)
#pragma unroll
      for (int r = 0; r < 8; ++r)
        h[r + 8 * half][n * 16 + nn] = acc[n][r] + cg_b1[n * 16 + nn];
  }
  __syncthreads();
  if (lane < 16) {
    float m = 0.f;
    for (int j = 0; j < 192; ++j) m += h[lane][j];
    m *= (1.0f / 192.0f);
    float v = 0.f;
    for (int j = 0; j < 192; ++j) { float dlt = h[lane][j] - m; v += dlt * dlt; }
    float rstd = rsqrtf(v * (1.0f / 192.0f) + EPS);
    float dot = cg_b2[0];
    for (int j = 0; j < 192; ++j) {
      float y = (h[lane][j] - m) * rstd * cg_g[j] + cg_beta[j];
      y = fmaxf(y, 0.f);
      dot += y * cg_W2[j];
    }
    gate[lane] = 1.0f / (1.0f + __expf(-dot));
  }
  __syncthreads();
  // gated output: xg*gate + glob*(1-gate), bf16 token-major
  for (int row = 0; row < 16; ++row) {
    float gt = gate[row];
    int t = t0 + row;
    const float* xr = x + (size_t)(t + b + 1) * DD;
    const float* gr = globF + (size_t)(grow0 + row) * DD;
    __bf16* orow = xgB + (size_t)t * DD;
    for (int d = lane; d < DD; d += 32)
      orow[d] = (__bf16)(xr[d] * gt + gr[d] * (1.0f - gt));
  }
}

// ---------------- conv over G axis (3 taps of DxD, WMMA) --------------------
// A rows (16 tokens x 384 per tap, contiguous) are staged to LDS with
// GLOBAL_LOAD_ASYNC_TO_LDS_B128 (ASYNCcnt) and shared by all 8 waves; each
// wave computes its own 16x16 output tile, transposed f32 store (feature-major)
__global__ __launch_bounds__(256) void k_conv(const __bf16* __restrict__ xgB,
                                              const __bf16* __restrict__ convWt,
                                              const float* __restrict__ conv_b,
                                              float* __restrict__ xc) {
  __shared__ __bf16 sA[3][16][DD];   // 36 KB
  int t0 = blockIdx.x * 16;
  int b = t0 / (GG * LL);
  int g = (t0 / LL) % GG;
  int l0 = t0 % LL;
  int tid = threadIdx.x;
  int wave = tid >> 5;
  int n0 = (blockIdx.y * 8 + wave) * 16;

  // wave-cluster path: NOP unless dispatched as a workgroup cluster
  if (__builtin_amdgcn_cluster_id_x() != 0) __builtin_amdgcn_s_cluster_barrier();

  // ---- async stage A tiles (one contiguous 12 KB copy per valid tap) ----
#pragma unroll
  for (int tap = 0; tap < 3; ++tap) {
    int gp = g + tap - 1;
    if ((unsigned)gp < (unsigned)GG) {
      int arow0 = t0 + (tap - 1) * LL;
      const char* gbase = (const char*)(xgB + (size_t)arow0 * DD);
      unsigned lbase = (unsigned)(uintptr_t)(&sA[tap][0][0]);  // LDS offset (flat-aperture low bits)
#pragma unroll
      for (int it = 0; it < 3; ++it) {
        int boff = (it * 256 + tid) * 16;          // 3*256*16 = 12288 bytes
        unsigned long long ga = (unsigned long long)(uintptr_t)(gbase + boff);
        unsigned la = lbase + boff;
        asm volatile("global_load_async_to_lds_b128 %0, %1, off"
                     :: "v"(la), "v"(ga) : "memory");
      }
    }
  }
  asm volatile("s_wait_asynccnt 0x0" ::: "memory");
  __syncthreads();

  // ---- WMMA main loop: A from LDS (shared across waves), B from VMEM ----
  v8f acc = (v8f)0.0f;
#pragma unroll
  for (int tap = 0; tap < 3; ++tap) {
    int gp = g + tap - 1;
    if ((unsigned)gp < (unsigned)GG) {
      const __bf16* wt = convWt + (size_t)tap * DD * DD;
#pragma unroll
      for (int k = 0; k < 12; ++k) {
        __builtin_prefetch(wt + (size_t)(n0) * DD + (k + 1) * 32, 0, 3);
        v16bf a = load_a_ldsb(&sA[tap][0][0], DD, k * 32);
        v16bf bm = load_b(wt, DD, k * 32, n0);
        acc = wmma_bf16(a, bm, acc);
      }
    }
  }
  int lane = tid & 31;
  int nn = lane & 15, half = lane >> 4;
  float bias = conv_b[n0 + nn];
  // transposed store: feature-major [b][g][d][l] -> FFT reads contiguous
  float* dst = xc + ((size_t)(b * GG + g) * DD + n0 + nn) * LL + l0 + 8 * half;
#pragma unroll
  for (int r = 0; r < 8; ++r) dst[r] = acc[r] + bias;
}

// ---------------- band-weight branch ----------------
__global__ __launch_bounds__(256) void k_bd_mean(const float* __restrict__ xc,
                                                 float* __restrict__ xmean) {
  int idx = blockIdx.x * 8 + (threadIdx.x >> 5);   // (b*G+g)*D + d
  int lane = threadIdx.x & 31;
  const float* s = xc + (size_t)idx * LL;
  float sum = 0.f;
  for (int i = lane; i < LL; i += 32) sum += s[i];
#pragma unroll
  for (int off = 16; off > 0; off >>= 1) sum += __shfl_xor(sum, off, 32);
  if (lane == 0) xmean[idx] = sum * (1.0f / (float)LL);
}

__global__ __launch_bounds__(96) void k_bd_mlp(const float* __restrict__ xmean,
                                               const float* __restrict__ W1,
                                               const float* __restrict__ b1,
                                               const float* __restrict__ gg,
                                               const float* __restrict__ beta,
                                               const float* __restrict__ W2,
                                               const float* __restrict__ b2,
                                               float* __restrict__ bw) {
  __shared__ float h[96];
  __shared__ float stat[2];
  __shared__ float outs[3];
  int bg = blockIdx.x;
  int j = threadIdx.x;
  const float* xm = xmean + (size_t)bg * DD;
  float a = b1[j];
  for (int d = 0; d < DD; ++d) a += xm[d] * W1[(size_t)d * HH + j];
  h[j] = a;
  __syncthreads();
  if (j == 0) {
    float m = 0.f;
    for (int i = 0; i < HH; ++i) m += h[i];
    m *= (1.0f / HH);
    float v = 0.f;
    for (int i = 0; i < HH; ++i) { float dlt = h[i] - m; v += dlt * dlt; }
    stat[0] = m;
    stat[1] = rsqrtf(v * (1.0f / HH) + EPS);
  }
  __syncthreads();
  float y = fmaxf((h[j] - stat[0]) * stat[1] * gg[j] + beta[j], 0.f);
  __syncthreads();
  h[j] = y;
  __syncthreads();
  if (j < 3) {
    float o = b2[j];
    for (int i = 0; i < HH; ++i) o += h[i] * W2[i * 3 + j];
    outs[j] = o;
  }
  __syncthreads();
  if (j == 0) {
    float mx = fmaxf(outs[0], fmaxf(outs[1], outs[2]));
    float e0 = __expf(outs[0] - mx), e1 = __expf(outs[1] - mx), e2 = __expf(outs[2] - mx);
    float inv = 1.0f / (e0 + e1 + e2);
    bw[bg * 3 + 0] = e0 * inv;
    bw[bg * 3 + 1] = e1 * inv;
    bw[bg * 3 + 2] = e2 * inv;
  }
}

__global__ __launch_bounds__(256) void k_wm(const float* __restrict__ bw, float* __restrict__ wm) {
  int i = blockIdx.x * 256 + threadIdx.x;
  if (i >= BB * GG * LL) return;
  int bg = i >> 10, l = i & 1023;
  int f = (l <= 512) ? l : (LL - l);
  int sel = (f <= LL / 8) ? 0 : ((f <= LL / 4) ? 1 : 2);
  wm[i] = bw[bg * 3 + sel];
}

// ---------------- FFT core (radix-2 DIT in LDS, 1024-pt, 128 threads) ----------------
__device__ __forceinline__ void fft1024(float* re, float* im, int tid) {
#pragma unroll
  for (int s = 0; s < 10; ++s) {
    int half = 1 << s;
    for (int t = tid; t < 512; t += 128) {
      int grp = t >> s;
      int pos = t & (half - 1);
      int i0 = (grp << (s + 1)) + pos;
      int i1 = i0 + half;
      float ang = -3.14159265358979323846f * (float)pos / (float)half;
      float wi, wr;
      __sincosf(ang, &wi, &wr);
      float xr = re[i1], xi = im[i1];
      float tr = wr * xr - wi * xi;
      float ti = wr * xi + wi * xr;
      float ur = re[i0], ui = im[i0];
      re[i1] = ur - tr; im[i1] = ui - ti;
      re[i0] = ur + tr; im[i0] = ui + ti;
    }
    __syncthreads();
  }
}

__global__ __launch_bounds__(128) void k_fft(const float* __restrict__ xc,
                                             float* __restrict__ magF,
                                             float* __restrict__ phF,
                                             __bf16* __restrict__ magB,
                                             __bf16* __restrict__ phB) {
  __shared__ float re[1024];
  __shared__ float im[1024];
  int bgd = blockIdx.x;
  int d = bgd % DD;
  int bg = bgd / DD;
  int tid = threadIdx.x;
  const float* src = xc + (size_t)bgd * LL;
  for (int i = tid; i < LL; i += 128) {
    int j = __brev((unsigned)i) >> 22;
    re[j] = src[i];
    im[j] = 0.f;
  }
  __syncthreads();
  fft1024(re, im, tid);
  for (int i = tid; i < LL; i += 128) {
    float rr = re[i], ii = im[i];
    float mg = sqrtf(rr * rr + ii * ii);
    float ph = atan2f(ii, rr);
    magF[(size_t)bgd * LL + i] = mg;
    phF[(size_t)bgd * LL + i] = ph;
    size_t tok = (size_t)(bg * LL + i) * DD + d;  // token-major for WMMA A loads
    magB[tok] = (__bf16)mg;
    phB[tok] = (__bf16)ph;
  }
}

// ---------------- spectral attention MLP (WMMA two stages, LN between) ----------------
__global__ __launch_bounds__(32) void k_att_mlp(const __bf16* __restrict__ AinB,
                                                const __bf16* __restrict__ W1t,
                                                const float* __restrict__ b1,
                                                const float* __restrict__ gg,
                                                const float* __restrict__ beta,
                                                const __bf16* __restrict__ W2t,
                                                const float* __restrict__ b2,
                                                float* __restrict__ OutT,
                                                int act /*0=sigmoid,1=tanh*/) {
  __shared__ float h[16][96];
  int t0 = blockIdx.x * 16;
  int bg = t0 / LL;
  int l0 = t0 % LL;
  int lane = threadIdx.x;

  // stage 1: (16 x 384) x (384 x 96)
  v8f acc[6];
#pragma unroll
  for (int n = 0; n < 6; ++n) acc[n] = (v8f)0.0f;
  for (int k = 0; k < 12; ++k) {
    v16bf a = load_a(AinB, DD, t0, k * 32);
#pragma unroll
    for (int n = 0; n < 6; ++n) {
      v16bf bm = load_b(W1t, DD, k * 32, n * 16);
      acc[n] = wmma_bf16(a, bm, acc[n]);
    }
  }
  {
    int nn = lane & 15, half = lane >> 4;
#pragma unroll
    for (int n = 0; n < 6; ++n)
#pragma unroll
      for (int r = 0; r < 8; ++r)
        h[r + 8 * half][n * 16 + nn] = acc[n][r] + b1[n * 16 + nn];
  }
  __syncthreads();
  if (lane < 16) {
    float m = 0.f;
    for (int j = 0; j < HH; ++j) m += h[lane][j];
    m *= (1.0f / HH);
    float v = 0.f;
    for (int j = 0; j < HH; ++j) { float dlt = h[lane][j] - m; v += dlt * dlt; }
    float rstd = rsqrtf(v * (1.0f / HH) + EPS);
    for (int j = 0; j < HH; ++j) {
      float y = (h[lane][j] - m) * rstd * gg[j] + beta[j];
      h[lane][j] = fmaxf(y, 0.f);
    }
  }
  __syncthreads();

  // stage 2: (16 x 96) x (96 x 384)
  v16bf af[3];
#pragma unroll
  for (int kk = 0; kk < 3; ++kk) af[kk] = load_a_lds(&h[0][0], HH, kk * 32);

  int nn = lane & 15, half = lane >> 4;
  for (int nt = 0; nt < 24; ++nt) {
    v8f a2 = (v8f)0.0f;
#pragma unroll
    for (int kk = 0; kk < 3; ++kk) {
      v16bf bm = load_b(W2t, HH, kk * 32, nt * 16);
      a2 = wmma_bf16(af[kk], bm, a2);
    }
    float bias = b2[nt * 16 + nn];
    float* dst = OutT + ((size_t)bg * DD + nt * 16 + nn) * LL + l0 + 8 * half;
#pragma unroll
    for (int r = 0; r < 8; ++r) {
      float v = a2[r] + bias;
      v = act ? tanhf(v) : (1.0f / (1.0f + __expf(-v)));
      dst[r] = v;
    }
  }
}

// ---------------- combine + inverse FFT + scatter to output ----------------
__global__ __launch_bounds__(128) void k_ifft(const float* __restrict__ magF,
                                              const float* __restrict__ phF,
                                              const float* __restrict__ matt,
                                              const float* __restrict__ patt,
                                              const float* __restrict__ wm,
                                              float* __restrict__ out) {
  __shared__ float re[1024];
  __shared__ float im[1024];
  int bgd = blockIdx.x;
  int d = bgd % DD;
  int bg = bgd / DD;
  int b = bg / GG;
  int tid = threadIdx.x;
  const float* mg = magF + (size_t)bgd * LL;
  const float* ph = phF + (size_t)bgd * LL;
  const float* ma = matt + (size_t)bgd * LL;
  const float* pa = patt + (size_t)bgd * LL;
  const float* w = wm + (size_t)bg * LL;
  for (int i = tid; i < LL; i += 128) {
    float wv = w[i];
    float m = mg[i] * wv * ma[i];
    float p = ph[i] * wv + pa[i];
    float s, c;
    __sincosf(p, &s, &c);
    int j = __brev((unsigned)i) >> 22;
    re[j] = m * c;
    im[j] = -(m * s);  // conjugate -> inverse via forward FFT
  }
  __syncthreads();
  fft1024(re, im, tid);
  for (int i = tid; i < LL; i += 128) {
    // real(ifft) = real(fft(conj(X))) / N ; scatter token-major into output
    out[((size_t)(bg * LL + i) + b + 1) * DD + d] = re[i] * (1.0f / (float)LL);
  }
}

__global__ __launch_bounds__(256) void k_cls(const float* __restrict__ x, float* __restrict__ out) {
  int i = blockIdx.x * 256 + threadIdx.x;
  if (i >= BB * DD) return;
  int b = i / DD, d = i % DD;
  size_t off = (size_t)b * (GG * LL + 1) * DD + d;
  out[off] = x[off];
}

// ---------------- host orchestration ----------------
extern "C" void kernel_launch(void* const* d_in, const int* in_sizes, int n_in,
                              void* d_out, int out_size, void* d_ws, size_t ws_size,
                              hipStream_t stream) {
  const float* x       = (const float*)d_in[0];
  const float* mg_W1   = (const float*)d_in[1];
  const float* mg_b1   = (const float*)d_in[2];
  const float* mg_g    = (const float*)d_in[3];
  const float* mg_beta = (const float*)d_in[4];
  const float* mg_W2   = (const float*)d_in[5];
  const float* mg_b2   = (const float*)d_in[6];
  const float* ph_W1   = (const float*)d_in[7];
  const float* ph_b1   = (const float*)d_in[8];
  const float* ph_g    = (const float*)d_in[9];
  const float* ph_beta = (const float*)d_in[10];
  const float* ph_W2   = (const float*)d_in[11];
  const float* ph_b2   = (const float*)d_in[12];
  const float* bd_W1   = (const float*)d_in[13];
  const float* bd_b1   = (const float*)d_in[14];
  const float* bd_g    = (const float*)d_in[15];
  const float* bd_beta = (const float*)d_in[16];
  const float* bd_W2   = (const float*)d_in[17];
  const float* bd_b2   = (const float*)d_in[18];
  const float* cg_W1   = (const float*)d_in[19];
  const float* cg_b1   = (const float*)d_in[20];
  const float* cg_g    = (const float*)d_in[21];
  const float* cg_beta = (const float*)d_in[22];
  const float* cg_W2   = (const float*)d_in[23];
  const float* cg_b2   = (const float*)d_in[24];
  const float* conv_W  = (const float*)d_in[25];
  const float* conv_b  = (const float*)d_in[26];
  float* out = (float*)d_out;

  // workspace carve-up
  size_t off = 0;
  auto alloc = [&](size_t bytes) -> void* {
    void* p = (char*)d_ws + off;
    off += (bytes + 255) & ~(size_t)255;
    return p;
  };
  const size_t TOKD = (size_t)TT * DD;      // 37,748,736
  const size_t BLD  = (size_t)BL * DD;      // 12,582,912

  __bf16* convWt = (__bf16*)alloc(3 * DD * DD * 2);
  __bf16* mgW1t  = (__bf16*)alloc((size_t)HH * DD * 2);
  __bf16* mgW2t  = (__bf16*)alloc((size_t)DD * HH * 2);
  __bf16* phW1t  = (__bf16*)alloc((size_t)HH * DD * 2);
  __bf16* phW2t  = (__bf16*)alloc((size_t)DD * HH * 2);
  __bf16* cgW1t  = (__bf16*)alloc((size_t)192 * 768 * 2);
  __bf16* xB     = (__bf16*)alloc(TOKD * 2);   // reused as magB after k_gate
  float*  globF  = (float*) alloc(BLD * 4);
  __bf16* globB  = (__bf16*)alloc(BLD * 2);
  __bf16* xgB    = (__bf16*)alloc(TOKD * 2);   // reused as phB after k_conv
  float*  xc     = (float*) alloc(TOKD * 4);   // reused as matt after k_fft
  float*  magF   = (float*) alloc(TOKD * 4);
  float*  phF    = (float*) alloc(TOKD * 4);
  float*  patt   = (float*) alloc(TOKD * 4);
  float*  xmean  = (float*) alloc((size_t)BB * GG * DD * 4);
  float*  bw     = (float*) alloc((size_t)BB * GG * 3 * 4);
  float*  wm     = (float*) alloc((size_t)BB * GG * LL * 4);
  __bf16* magB = xB;    // alias: xB dead after k_gate
  __bf16* phB  = xgB;   // alias: xgB dead after k_conv
  float*  matt = xc;    // alias: xc dead after k_fft / k_bd_mean
  (void)ws_size; (void)in_sizes; (void)n_in; (void)out_size;

  // 1) weight prep
  k_conv_w<<<(3 * DD * DD + 255) / 256, 256, 0, stream>>>(conv_W, convWt);
  k_transpose_bf16<<<(DD * HH + 255) / 256, 256, 0, stream>>>(mg_W1, mgW1t, DD, HH);
  k_transpose_bf16<<<(HH * DD + 255) / 256, 256, 0, stream>>>(mg_W2, mgW2t, HH, DD);
  k_transpose_bf16<<<(DD * HH + 255) / 256, 256, 0, stream>>>(ph_W1, phW1t, DD, HH);
  k_transpose_bf16<<<(HH * DD + 255) / 256, 256, 0, stream>>>(ph_W2, phW2t, HH, DD);
  k_transpose_bf16<<<(768 * 192 + 255) / 256, 256, 0, stream>>>(cg_W1, cgW1t, 768, 192);

  // 2) glob mean + bf16 casts
  k_glob<<<(BLD + 255) / 256, 256, 0, stream>>>(x, globF, globB, xB);

  // 3) gate MLP (WMMA) + gating
  k_gate<<<TT / 16, 32, 0, stream>>>(x, globF, xB, globB, cgW1t,
                                     cg_b1, cg_g, cg_beta, cg_W2, cg_b2, xgB);

  // 4) conv across G (WMMA, async-LDS staged A), transposed fp32 store
  k_conv<<<dim3(TT / 16, 3, 1), 256, 0, stream>>>(xgB, convWt, conv_b, xc);

  // 5) band-weight branch
  k_bd_mean<<<(BB * GG * DD) / 8, 256, 0, stream>>>(xc, xmean);
  k_bd_mlp<<<BB * GG, 96, 0, stream>>>(xmean, bd_W1, bd_b1, bd_g, bd_beta, bd_W2, bd_b2, bw);
  k_wm<<<(BB * GG * LL) / 256, 256, 0, stream>>>(bw, wm);

  // 6) forward FFT -> mag/phase (f32 feature-major + bf16 token-major)
  k_fft<<<BB * GG * DD, 128, 0, stream>>>(xc, magF, phF, magB, phB);

  // 7) spectral attention MLPs (WMMA)
  k_att_mlp<<<TT / 16, 32, 0, stream>>>(magB, mgW1t, mg_b1, mg_g, mg_beta, mgW2t, mg_b2, matt, 0);
  k_att_mlp<<<TT / 16, 32, 0, stream>>>(phB, phW1t, ph_b1, ph_g, ph_beta, phW2t, ph_b2, patt, 1);

  // 8) combine + inverse FFT + scatter
  k_ifft<<<BB * GG * DD, 128, 0, stream>>>(magF, phF, matt, patt, wm, out);

  // 9) cls passthrough
  k_cls<<<(BB * DD + 255) / 256, 256, 0, stream>>>(x, out);
}
